// TransformerBlock_84370337563121
// MI455X (gfx1250) — matrix operand order repaired and checked by reading, not compiled, not run on hardware
//
#include <hip/hip_runtime.h>

#define Bn 4
#define Hn 16
#define Sn 2048
#define Dn 64

typedef _Float16 v16h __attribute__((ext_vector_type(16)));
typedef _Float16 v8h  __attribute__((ext_vector_type(8)));
typedef _Float16 v4h  __attribute__((ext_vector_type(4)));
typedef __fp16   v2fp __attribute__((ext_vector_type(2)));
typedef float    v8f  __attribute__((ext_vector_type(8)));
typedef float    v4f  __attribute__((ext_vector_type(4)));

// D^-0.5 * log2(e): scores land directly in base-2 domain -> bare v_exp_f32.
#define QSCALE 0.18033688011112042f

static __device__ __forceinline__ v8f wmma_f16(v16h a, v16h b, v8f c) {
  // D = A(16x32 f16) x B(32x16 f16) + C(16x16 f32)
  return __builtin_amdgcn_wmma_f32_16x16x32_f16(false, a, false, b, (short)0, c,
                                                false, false);
}

static __device__ __forceinline__ unsigned pkh2(float a, float b) {
  union { v2fp h; unsigned u; } u;
  u.h = __builtin_amdgcn_cvt_pkrtz(a, b);  // v_cvt_pk_rtz_f16_f32
  return u.u;
}

union U16 { v16h v; v8h h[2]; unsigned u[8]; };

// Flash attention forward, causal. One wave owns 16 query rows x full D=64;
// block = 4 waves -> 64 query rows. KV staged in LDS as f16 per 32-key tile.
// S^T = K * Q^T so softmax stats are per-lane (q = lane%16); the exp'd S^T
// accumulator, after a lane-half exchange, gives the B operand for O^T = V^T * P^T, so the
// running rescale (alpha) and final 1/l are per-lane scalars.
__global__ __launch_bounds__(128)
void fa_fwd_causal(const float* __restrict__ Qg, const float* __restrict__ Kg,
                   const float* __restrict__ Vg, float* __restrict__ Og) {
  __shared__ _Float16 sK[32 * 72];  // K tile row-major [kv][d], pad 64->72
  __shared__ _Float16 sV[64 * 40];  // V^T tile row-major [d][kv], pad 32->40

  const int tid  = threadIdx.x;
  const int lane = tid & 31;
  const int wv   = tid >> 5;
  const int hi   = lane >> 4;   // which 16-lane half
  const int ln   = lane & 15;
  const int aoff = hi << 3;     // S^T accumulator row base: element v holds kv aoff + v (+16 in st1)

  const int qt = blockIdx.x & 31;        // 32 query tiles of 64 rows
  const int bh = blockIdx.x >> 5;        // 64 (b,h) pairs
  const int q0 = qt * 64;
  const int qw = q0 + wv * 16;           // this wave's first query row
  const int qrow = qw + ln;              // this lane's query row (dup in both halves)

  const size_t base = (size_t)bh * Sn * Dn;
  const float* qp = Qg + base;
  const float* kp = Kg + base;
  const float* vp = Vg + base;
  float* op = Og + base;

  // Q^T as B operand: lane holds column q=ln, K(=d) = 16*hi + 0..15 (+32c).
  v16h Qb[2];
#pragma unroll
  for (int c = 0; c < 2; ++c) {
    const float* src = qp + (size_t)qrow * Dn + 32 * c + 16 * hi;
#pragma unroll
    for (int j = 0; j < 16; ++j)
      Qb[c][j] = (_Float16)(src[j] * QSCALE);
  }

  const v8f vzero = {0, 0, 0, 0, 0, 0, 0, 0};
  v8f Oc[4] = {vzero, vzero, vzero, vzero};  // O^T chunks: d = 16c + v + 8*hi, q = ln
  float m = -1e30f, l = 0.f;  // base-2 domain

  // Cooperative loader geometry: each thread owns a 4-row x 4-col block.
  const int lr4 = (tid >> 4) << 2;  // kv rows lr4..lr4+3
  const int ld4 = (tid & 15) << 2;  // d cols  ld4..ld4+3

  const int nt = (q0 >> 5) + 2;  // kv tiles of 32, up to q0+63
  for (int t = 0; t < nt; ++t) {
    const int kv0 = t << 5;

    __syncthreads();
    {
      v4f kr[4], vr[4];
#pragma unroll
      for (int rr = 0; rr < 4; ++rr) {
        kr[rr] = *(const v4f*)(kp + (size_t)(kv0 + lr4 + rr) * Dn + ld4);
        vr[rr] = *(const v4f*)(vp + (size_t)(kv0 + lr4 + rr) * Dn + ld4);
      }
#pragma unroll
      for (int rr = 0; rr < 4; ++rr) {
        v4h kh = {(_Float16)kr[rr][0], (_Float16)kr[rr][1],
                  (_Float16)kr[rr][2], (_Float16)kr[rr][3]};
        *(v4h*)(&sK[(lr4 + rr) * 72 + ld4]) = kh;
      }
#pragma unroll
      for (int i = 0; i < 4; ++i) {  // transpose: pack 4 consecutive kv per d
        v4h vh = {(_Float16)vr[0][i], (_Float16)vr[1][i],
                  (_Float16)vr[2][i], (_Float16)vr[3][i]};
        *(v4h*)(&sV[(ld4 + i) * 40 + lr4]) = vh;
      }
    }
    __syncthreads();

    if (kv0 > qw + 15) continue;  // wave-uniform: tile entirely above the diagonal

    // ---- S^T = K * Q^T : two 16x16 tiles (kv 0..15 and 16..31), K-dim = d = 64
    v8f st0 = vzero, st1 = vzero;
#pragma unroll
    for (int c = 0; c < 2; ++c) {
      U16 a0, a1;
      a0.h[0] = *(const v8h*)(&sK[ln * 72 + 32 * c + (hi << 4)]);
      a0.h[1] = *(const v8h*)(&sK[ln * 72 + 32 * c + (hi << 4) + 8]);
      a1.h[0] = *(const v8h*)(&sK[(16 + ln) * 72 + 32 * c + (hi << 4)]);
      a1.h[1] = *(const v8h*)(&sK[(16 + ln) * 72 + 32 * c + (hi << 4) + 8]);
      st0 = wmma_f16(a0.v, Qb[c], st0);
      st1 = wmma_f16(a1.v, Qb[c], st1);
    }

    // ---- causal mask (diagonal tiles only) + running max
    float tmax;
    if (kv0 + 32 > qw) {  // wave-uniform: tile touches the diagonal
      tmax = -INFINITY;
#pragma unroll
      for (int v = 0; v < 8; ++v) {
        const int kva = kv0 + aoff + v;
        float a = (kva > qrow) ? -INFINITY : (float)st0[v];
        float b = (kva + 16 > qrow) ? -INFINITY : (float)st1[v];
        st0[v] = a;
        st1[v] = b;
        tmax = fmaxf(tmax, fmaxf(a, b));
      }
    } else {
      tmax = fmaxf((float)st0[0], (float)st1[0]);
#pragma unroll
      for (int v = 1; v < 8; ++v)
        tmax = fmaxf(tmax, fmaxf((float)st0[v], (float)st1[v]));
    }
    tmax = fmaxf(tmax, __shfl_xor(tmax, 16));  // halves hold complementary kv sets
    const float mnew  = fmaxf(m, tmax);
    const float alpha = __builtin_amdgcn_exp2f(m - mnew);
    float rsum = 0.f;
    unsigned pk0[4], pk1[4];
#pragma unroll
    for (int j = 0; j < 4; ++j) {
      const float p00 = __builtin_amdgcn_exp2f((float)st0[2 * j]     - mnew);
      const float p01 = __builtin_amdgcn_exp2f((float)st0[2 * j + 1] - mnew);
      const float p10 = __builtin_amdgcn_exp2f((float)st1[2 * j]     - mnew);
      const float p11 = __builtin_amdgcn_exp2f((float)st1[2 * j + 1] - mnew);
      rsum += (p00 + p01) + (p10 + p11);
      pk0[j] = pkh2(p00, p01);  // halves kv = kv0 + aoff + {2j, 2j+1}
      pk1[j] = pkh2(p10, p11);  // halves kv = kv0 + 16 + aoff + {2j, 2j+1}
    }
    rsum += __shfl_xor(rsum, 16);
    l = l * alpha + rsum;
    m = mnew;

    // ---- assemble B = P^T: lane needs contiguous kv = 16*hi + 0..15
    // hi=0: [own t0 (kv 0..7) | partner t0 (kv 8..15)]
    // hi=1: [partner t1 (kv 16..23) | own t1 (kv 24..31)]
    U16 bp;
#pragma unroll
    for (int j = 0; j < 4; ++j) {
      const unsigned o0 = (unsigned)__shfl_xor((int)pk0[j], 16);
      const unsigned o1 = (unsigned)__shfl_xor((int)pk1[j], 16);
      bp.u[j]     = (hi == 0) ? pk0[j] : o1;
      bp.u[4 + j] = (hi == 0) ? o0     : pk1[j];
    }

    // ---- O^T += V^T * P^T, with running rescale by alpha (per-lane scalar)
#pragma unroll
    for (int c = 0; c < 4; ++c) {
      U16 av;  // A = V^T rows: M = d-16c = ln, K = kv 16*hi + 0..15
      av.h[0] = *(const v8h*)(&sV[(16 * c + ln) * 40 + (hi << 4)]);
      av.h[1] = *(const v8h*)(&sV[(16 * c + ln) * 40 + (hi << 4) + 8]);
      Oc[c] = Oc[c] * alpha;
      Oc[c] = wmma_f16(av.v, bp.v, Oc[c]);
    }
  }

  // ---- finalize: O[q][d] = Oc / l ; lane stores q=qrow, d = 16c + 8*hi + 0..7
  const float inv = 1.0f / l;
#pragma unroll
  for (int c = 0; c < 4; ++c) {
    float* dst = op + (size_t)qrow * Dn + 16 * c + (hi << 3);
    v4f lo = {Oc[c][0] * inv, Oc[c][1] * inv, Oc[c][2] * inv, Oc[c][3] * inv};
    v4f hh = {Oc[c][4] * inv, Oc[c][5] * inv, Oc[c][6] * inv, Oc[c][7] * inv};
    *(v4f*)dst = lo;
    *(v4f*)(dst + 4) = hh;
  }
}

extern "C" void kernel_launch(void* const* d_in, const int* in_sizes, int n_in,
                              void* d_out, int out_size, void* d_ws, size_t ws_size,
                              hipStream_t stream) {
  const float* q = (const float*)d_in[0];
  const float* k = (const float*)d_in[1];
  const float* v = (const float*)d_in[2];
  float* o = (float*)d_out;
  (void)in_sizes; (void)n_in; (void)out_size; (void)d_ws; (void)ws_size;
  dim3 grid(Bn * Hn * (Sn / 64));  // 2048 blocks: (b,h) x 32 query tiles
  fa_fwd_causal<<<grid, 128, 0, stream>>>(q, k, v, o);
}
